// CommNet_27187142983845
// MI455X (gfx1250) — matrix operand order repaired and checked by reading, not compiled, not run on hardware
//
#include <hip/hip_runtime.h>

// ---------------------------------------------------------------------------
// CommNet forward, MI455X (gfx1250, wave32, WMMA bf16 -> f32 acc, TDM staging).
// Shapes: b=16, n=128, obs=128, hid=256, act=32.  M = b*n = 2048 rows.
// Comm term folded into weights: W2 = f_W - C_W/127; per-batch bias
// T_b = (colsum(h)/127) @ C_W + f_b + C_b.  One WMMA GEMM per comm iter.
// GEMM: block = 128 thr (4 waves), tile 32 rows x 256 cols, K-step 32,
// A tile DMA'd to LDS once per block via tensor_load_to_lds with row padding
// (pad_interval=512B, pad_amount=1 dword -> 516B row stride, bank-skewed).
// ---------------------------------------------------------------------------

typedef __attribute__((ext_vector_type(16))) __bf16 v16bf;
typedef __attribute__((ext_vector_type(8)))  float  v8f;
typedef __attribute__((ext_vector_type(4)))  unsigned int u32x4;
typedef __attribute__((ext_vector_type(8)))  int          i32x8;
typedef __attribute__((ext_vector_type(4)))  int          i32x4;

union Frag {
    v16bf v;
    unsigned int u[8];
};

static __device__ __forceinline__ __bf16 f2bf(float x) { return (__bf16)x; }

// Pack two source elements into one dword of two bf16 (A-fragment pair).
static __device__ __forceinline__ unsigned int load_pair_bf16(const float* fp) {
    union { __bf16 b[2]; unsigned int u; } p;
    p.b[0] = (__bf16)fp[0];
    p.b[1] = (__bf16)fp[1];
    return p.u;
}
static __device__ __forceinline__ unsigned int load_pair_bf16(const __bf16* bp) {
    return *(const unsigned int*)bp;
}

// --------------------------- TDM availability ------------------------------
#if defined(__has_builtin)
#if __has_builtin(__builtin_amdgcn_tensor_load_to_lds) && \
    __has_builtin(__builtin_amdgcn_s_wait_tensorcnt)
#define USE_TDM 1
#endif
#endif
#ifndef USE_TDM
#define USE_TDM 0
#endif

#if USE_TDM
// Issue a 2D TDM load: tile (32 rows x tile0 elems) of a row-major tensor
// (row stride = dim0 elems) into LDS at ldsAddr, padding 1 dword per 512B row.
static __device__ __forceinline__ void tdm_load_tile(
    unsigned int ldsAddr, unsigned long long gaddr,
    int dataSizeCode, int dim0, int tile0)
{
    u32x4 g0;
    g0[0] = 1u;                                          // count=1, user mode
    g0[1] = ldsAddr;                                     // lds_addr [63:32]
    g0[2] = (unsigned int)(gaddr & 0xFFFFFFFFull);       // global_addr lo
    g0[3] = ((unsigned int)((gaddr >> 32) & 0x01FFFFFFull))
          | (2u << 30);                                  // addr hi | type=2
    i32x8 g1;
    g1[0] = (dataSizeCode << 16)                         // data_size
          | (1 << 20)                                    // pad_enable
          | (6 << 22);                                   // pad_interval: 512B
                                                         // pad_amount=0 -> 1 dword
    g1[1] = (dim0 & 0xFFFF) << 16;                       // tensor_dim0 lo16
    g1[2] = ((dim0 >> 16) & 0xFFFF) | ((2048 & 0xFFFF) << 16); // dim0 hi | dim1 lo
    g1[3] = (tile0 & 0xFFFF) << 16;                      // dim1 hi=0 | tile_dim0
    g1[4] = 32;                                          // tile_dim1=32, tile_dim2=0
    g1[5] = dim0;                                        // tensor_dim0_stride lo32
    g1[6] = 0;                                           // stride hi | dim1_stride lo
    g1[7] = 0;
    i32x4 g2 = {0, 0, 0, 0};
    i32x4 g3 = {0, 0, 0, 0};
#if __clang_major__ >= 23
    i32x8 g4 = {0, 0, 0, 0, 0, 0, 0, 0};
    __builtin_amdgcn_tensor_load_to_lds(g0, g1, g2, g3, g4, 0);
#else
    __builtin_amdgcn_tensor_load_to_lds(g0, g1, g2, g3, 0);
#endif
}
#endif

// ---------------------------------------------------------------------------
// prep: transpose + bf16-convert weights.
// ---------------------------------------------------------------------------
__global__ __launch_bounds__(256) void prep_k(
    const float* __restrict__ enc_W, const float* __restrict__ f_W,
    const float* __restrict__ C_W,   const float* __restrict__ f_b,
    const float* __restrict__ C_b,
    __bf16* __restrict__ encWt, __bf16* __restrict__ fWt,
    __bf16* __restrict__ W2t,   float* __restrict__ fbcb)
{
    int idx = blockIdx.x * 256 + threadIdx.x;
    if (idx < 256 * 256) {
        int k = idx >> 8, n = idx & 255;
        float fw = f_W[k * 256 + n];
        float cw = C_W[k * 256 + n];
        fWt[n * 256 + k] = f2bf(fw);
        W2t[n * 256 + k] = f2bf(fw - cw * (1.0f / 127.0f));
        if (k < 128) encWt[n * 128 + k] = f2bf(enc_W[k * 256 + n]);
    } else if (idx < 256 * 256 + 256) {
        int n = idx - 256 * 256;
        fbcb[n] = f_b[n] + C_b[n];
    }
}

// ---------------------------------------------------------------------------
// comm_bias: T[b,n] = (colsum_a h[b,a,:] / 127) @ C_W + f_b + C_b
// ---------------------------------------------------------------------------
__global__ __launch_bounds__(256) void comm_bias_k(
    const __bf16* __restrict__ h, const float* __restrict__ C_W,
    const float* __restrict__ f_b, const float* __restrict__ C_b,
    float* __restrict__ T)
{
    __shared__ float s[256];
    const int b = blockIdx.x;
    const int n = threadIdx.x;
    float acc = 0.0f;
    for (int a = 0; a < 128; ++a)
        acc += (float)h[(b * 128 + a) * 256 + n];
    s[n] = acc * (1.0f / 127.0f);
    __syncthreads();
    float t = f_b[n] + C_b[n];
    for (int k = 0; k < 256; ++k)
        t += s[k] * C_W[k * 256 + n];
    T[b * 256 + n] = t;
}

// ---------------------------------------------------------------------------
// WMMA GEMM + bias + tanh:  Out[M,256] = tanh(A[M,KD] @ W + bias[batch,:])
//   Wt transposed bf16 [256][KD].  Block: 4 waves, tile 32 rows x 256 cols.
//   Wave: 2 M-tiles x 4 N-tiles -> 8 wmma / K-step.  A tile staged ONCE in
//   LDS (516B skewed rows); K loop is barrier-free.
// ---------------------------------------------------------------------------
template <typename AT, int KD>
__global__ __launch_bounds__(128) void gemm_tanh_k(
    const AT* __restrict__ A, const __bf16* __restrict__ Wt,
    const float* __restrict__ bias, int biasStride,
    __bf16* __restrict__ Out)
{
    // 32 rows x 512B data + 4B pad per row.
    __shared__ unsigned char Abuf[32 * 516 + 16];

    const int lane  = threadIdx.x & 31;
    const int wave  = threadIdx.x >> 5;
    const int rbase = blockIdx.x * 32;
    const int wbase = wave * 64;

    const int rowA = lane & 15;
    const int khiA = (lane >= 16) ? 16 : 0;   // A-frag K offset for hi lanes
    const int khiB = (lane >= 16) ? 16 : 0;   // B-frag K offset for hi lanes
    const int colA = lane & 15;

    // ---- Stage A block-tile (32 x KD) into LDS with 516B row stride ----
#if USE_TDM
    if (threadIdx.x < 32) {
        tdm_load_tile((unsigned int)(unsigned long long)(uintptr_t)Abuf,
                      (unsigned long long)(uintptr_t)(A + rbase * KD),
                      (sizeof(AT) == 4) ? 2 : 1, KD, KD);
        __builtin_amdgcn_s_wait_tensorcnt(0);
    }
    __syncthreads();
#else
    {
        const unsigned int* src = (const unsigned int*)(A + rbase * KD);
        const int nwords = (32 * KD * (int)sizeof(AT)) / 4;
        for (int j = threadIdx.x; j < nwords; j += 128) {
            int byteoff = j * 4;
            int row     = byteoff >> 9;
            int within  = byteoff & 511;
            *(unsigned int*)(Abuf + row * 516 + within) = src[j];
        }
    }
    __syncthreads();
#endif

    v8f acc[2][4] = {};

    for (int kb = 0; kb < KD; kb += 32) {
        // A fragments (2 M-tiles) from skewed LDS rows.
        Frag a[2];
        #pragma unroll
        for (int mt = 0; mt < 2; ++mt) {
            const unsigned char* rp = Abuf + (mt * 16 + rowA) * 516;
            #pragma unroll
            for (int v = 0; v < 8; ++v) {
                int koff = kb + v * 2 + khiA;
                a[mt].u[v] = load_pair_bf16(
                    (const AT*)(rp + koff * (int)sizeof(AT)));
            }
        }

        if (kb + 32 < KD)
            __builtin_prefetch(&Wt[(wbase + colA) * KD + kb + 32 + khiB], 0, 1);

        #pragma unroll
        for (int t = 0; t < 4; ++t) {
            // B fragment: 32x16 bf16, lane = col, VGPR v = K pair 2v (+16 hi).
            const __bf16* wp = Wt + (wbase + t * 16 + colA) * KD + kb + khiB;
            Frag b;
            #pragma unroll
            for (int v = 0; v < 8; ++v)
                b.u[v] = *(const unsigned int*)&wp[2 * v];

            #pragma unroll
            for (int mt = 0; mt < 2; ++mt)
                acc[mt][t] = __builtin_amdgcn_wmma_f32_16x16x32_bf16(
                    false, a[mt].v, false, b.v, (short)0, acc[mt][t],
                    false, false);
        }
    }

    // Epilogue: C/D layout — VGPR r: lanes0-15 -> M=r, lanes16-31 -> M=r+8.
    const int mbase = (lane >= 16) ? 8 : 0;
    const int batch = rbase >> 7;            // 32-row tile never straddles a batch
    #pragma unroll
    for (int mt = 0; mt < 2; ++mt) {
        #pragma unroll
        for (int t = 0; t < 4; ++t) {
            int col  = wbase + t * 16 + colA;
            float bb = bias[batch * biasStride + col];
            #pragma unroll
            for (int r = 0; r < 8; ++r) {
                int row = rbase + mt * 16 + mbase + r;
                Out[row * 256 + col] = f2bf(tanhf(acc[mt][t][r] + bb));
            }
        }
    }
}

// ---------------------------------------------------------------------------
// head: one wave32 per row; lane == action column (act_dim == 32):
//   action = log_softmax(h @ act_W + act_b); value = h @ val_W + val_b
// ---------------------------------------------------------------------------
__global__ __launch_bounds__(128) void head_k(
    const __bf16* __restrict__ h,
    const float* __restrict__ act_W, const float* __restrict__ act_b,
    const float* __restrict__ val_W, const float* __restrict__ val_b,
    float* __restrict__ outA, float* __restrict__ outV)
{
    const int lane = threadIdx.x & 31;
    const int r    = blockIdx.x * 4 + (threadIdx.x >> 5);
    const __bf16* hr = h + r * 256;

    float acc  = act_b[lane];
    float vacc = 0.0f;
    for (int k = 0; k < 256; ++k)
        acc += (float)hr[k] * act_W[k * 32 + lane];
    for (int k = lane; k < 256; k += 32)
        vacc += (float)hr[k] * val_W[k];

    float m = acc;
    for (int off = 16; off >= 1; off >>= 1)
        m = fmaxf(m, __shfl_xor(m, off, 32));
    float e = __expf(acc - m);
    float se = e;
    for (int off = 16; off >= 1; off >>= 1)
        se += __shfl_xor(se, off, 32);
    float lse = m + __logf(se);
    outA[r * 32 + lane] = acc - lse;

    for (int off = 16; off >= 1; off >>= 1)
        vacc += __shfl_xor(vacc, off, 32);
    if (lane == 0) outV[r] = vacc + val_b[0];
}

// ---------------------------------------------------------------------------
extern "C" void kernel_launch(void* const* d_in, const int* in_sizes, int n_in,
                              void* d_out, int out_size, void* d_ws, size_t ws_size,
                              hipStream_t stream) {
    const float* obs   = (const float*)d_in[0];
    const float* enc_W = (const float*)d_in[1];
    const float* enc_b = (const float*)d_in[2];
    const float* f_W   = (const float*)d_in[3];
    const float* f_b   = (const float*)d_in[4];
    const float* C_W   = (const float*)d_in[5];
    const float* C_b   = (const float*)d_in[6];
    const float* act_W = (const float*)d_in[7];
    const float* act_b = (const float*)d_in[8];
    const float* val_W = (const float*)d_in[9];
    const float* val_b = (const float*)d_in[10];

    char* ws = (char*)d_ws;
    __bf16* encWt = (__bf16*)(ws);                                     //  64 KB
    __bf16* fWt   = (__bf16*)(ws + 65536);                             // 128 KB
    __bf16* W2t   = (__bf16*)(ws + 65536 + 131072);                    // 128 KB
    float*  fbcb  = (float*) (ws + 65536 + 2 * 131072);                //   1 KB
    float*  T     = (float*) (ws + 65536 + 2 * 131072 + 1024);         //  16 KB
    __bf16* hA    = (__bf16*)(ws + 65536 + 2 * 131072 + 1024 + 16384); //   1 MB
    __bf16* hB    = hA + 2048 * 256;                                   //   1 MB

    float* outA = (float*)d_out;          // [16,128,32]
    float* outV = outA + 16 * 128 * 32;   // [16,128,1]

    // Weight prep (transpose + bf16 + W2 fold + fused bias).
    prep_k<<<257, 256, 0, stream>>>(enc_W, f_W, C_W, f_b, C_b,
                                    encWt, fWt, W2t, fbcb);

    // Encoder: h = tanh(obs @ enc_W + enc_b)   (K=128, bias broadcast)
    gemm_tanh_k<float, 128><<<64, 128, 0, stream>>>(obs, encWt, enc_b, 0, hA);

    // Comm iter 0 (c == 0): h = tanh(h @ f_W + f_b + C_b)
    gemm_tanh_k<__bf16, 256><<<64, 128, 0, stream>>>(hA, fWt, fbcb, 0, hB);

    // Comm iters 1..3: T_b = (colsum/127)@C_W + f_b + C_b; h = tanh(h@W2 + T_b)
    const __bf16* cur = hB;
    __bf16* nxt = hA;
    for (int i = 1; i < 4; ++i) {
        comm_bias_k<<<16, 256, 0, stream>>>(cur, C_W, f_b, C_b, T);
        gemm_tanh_k<__bf16, 256><<<64, 128, 0, stream>>>(cur, W2t, T, 256, nxt);
        const __bf16* tmp = cur; cur = nxt; nxt = (__bf16*)tmp;
    }
    // Final h lives in hA (== cur).

    head_k<<<512, 128, 0, stream>>>(cur, act_W, act_b, val_W, val_b, outA, outV);
}
